// EntropyOptimizedLinear_10720238370983
// MI455X (gfx1250) — matrix operand order repaired and checked by reading, not compiled, not run on hardware
//
#include <hip/hip_runtime.h>
#include <hip/hip_fp16.h>

typedef _Float16 v4h  __attribute__((ext_vector_type(4)));
typedef _Float16 v8h  __attribute__((ext_vector_type(8)));
typedef _Float16 v16h __attribute__((ext_vector_type(16)));
typedef float    v2f  __attribute__((ext_vector_type(2)));
typedef float    v8f  __attribute__((ext_vector_type(8)));

constexpr int M_DIM = 16384;   // batch
constexpr int N_DIM = 512;     // out features
constexpr int K_DIM = 2048;    // in features

constexpr int NUM_BINS = 256;
constexpr float ENTROPY_THRESHOLD = 0.1f;

// ---------------------------------------------------------------------------
// Workspace init (harness poisons ws with 0xAA; we must zero our accumulator)
// ---------------------------------------------------------------------------
__global__ void ws_init_kernel(float* wsf) {
  if (threadIdx.x == 0 && blockIdx.x == 0) wsf[0] = 0.0f;
}

// ---------------------------------------------------------------------------
// Per-sample entropy: one workgroup (256 threads = 8 wave32) per row.
// min/max reduce -> LDS histogram -> Shannon entropy -> scaling -> atomicAdd.
// ---------------------------------------------------------------------------
__global__ __launch_bounds__(256) void entropy_kernel(const float* __restrict__ x,
                                                      float* __restrict__ sum_scale) {
  __shared__ float red[256];
  __shared__ unsigned bins[NUM_BINS];
  const int tid = threadIdx.x;
  const float* row = x + (size_t)blockIdx.x * K_DIM;

  float mn = 3.402823466e38f, mx = -3.402823466e38f;
  for (int i = tid; i < K_DIM; i += 256) {
    float v = row[i];
    mn = fminf(mn, v);
    mx = fmaxf(mx, v);
  }
  red[tid] = mn; __syncthreads();
  for (int s = 128; s > 0; s >>= 1) { if (tid < s) red[tid] = fminf(red[tid], red[tid + s]); __syncthreads(); }
  mn = red[0]; __syncthreads();
  red[tid] = mx; __syncthreads();
  for (int s = 128; s > 0; s >>= 1) { if (tid < s) red[tid] = fmaxf(red[tid], red[tid + s]); __syncthreads(); }
  mx = red[0]; __syncthreads();

  const float inv_rng = 1.0f / fmaxf(mx - mn, 1e-12f);
  bins[tid] = 0u;                 // 256 threads == NUM_BINS
  __syncthreads();
  for (int i = tid; i < K_DIM; i += 256) {
    int idx = (int)((row[i] - mn) * inv_rng * (float)NUM_BINS);
    idx = idx < 0 ? 0 : (idx > NUM_BINS - 1 ? NUM_BINS - 1 : idx);
    atomicAdd(&bins[idx], 1u);
  }
  __syncthreads();
  float p = (float)bins[tid] * (1.0f / (float)K_DIM);
  red[tid] = (p > 0.0f) ? -p * logf(p) : 0.0f;
  __syncthreads();
  for (int s = 128; s > 0; s >>= 1) { if (tid < s) red[tid] += red[tid + s]; __syncthreads(); }
  if (tid == 0) {
    float ent = red[0] * (1.0f / logf((float)NUM_BINS));   // normalized [0,1]
    float scale = fminf(ent * (1.0f / ENTROPY_THRESHOLD), 1.0f);
    atomicAdd(sum_scale, scale);
  }
}

__global__ void finalize_kernel(const float* wsf, int* flag) {
  if (threadIdx.x == 0 && blockIdx.x == 0) {
    float avg = wsf[0] * (1.0f / (float)M_DIM);
    flag[0] = (avg < 0.5f) ? 1 : 0;   // 1 -> half precision path
  }
}

// ---------------------------------------------------------------------------
// GEMM tiling shared by both precision paths:
//   block tile 128x128, 256 threads = 8 waves in 4(M) x 2(N) grid,
//   each wave owns a 32x64 tile = 2x4 WMMA 16x16 accumulators.
// ---------------------------------------------------------------------------
constexpr int BM = 128, BN = 128;

// ============================ fp16 WMMA path ===============================
constexpr int BK_H  = 32;
constexpr int LDA_H = BK_H + 8;   // 40-half row stride: conflict-free banks, 16B aligned

__global__ __launch_bounds__(256) void gemm_f16_wmma(const float* __restrict__ X,
                                                     const float* __restrict__ W,
                                                     const float* __restrict__ bias,
                                                     float* __restrict__ Y,
                                                     const int* __restrict__ flag) {
  if (flag[0] != 1) return;   // uniform across grid

  __shared__ alignas(16) _Float16 As[2][BM][LDA_H];
  __shared__ alignas(16) _Float16 Bs[2][BN][LDA_H];

  const int tid  = threadIdx.x;
  const int lane = tid & 31;
  const int wave = tid >> 5;
  const int wm   = wave & 3;     // M offset = wm*32
  const int wn   = wave >> 2;    // N offset = wn*64
  const int hs   = lane >> 4;    // lane half select (WMMA K split)
  const int l15  = lane & 15;
  const int bm   = blockIdx.x * BM;
  const int bn   = blockIdx.y * BN;

  v8f acc[2][4];
#pragma unroll
  for (int i = 0; i < 2; ++i)
#pragma unroll
    for (int j = 0; j < 4; ++j) acc[i][j] = v8f{};

  auto load_tiles = [&](int buf, int kk) {
#pragma unroll
    for (int i = 0; i < 4; ++i) {
      int chunk = tid + i * 256;        // 1024 float4 chunks per tile
      int r = chunk >> 3;               // 0..127
      int c = (chunk & 7) * 4;          // 0..28
      float4 va = *(const float4*)(X + (size_t)(bm + r) * K_DIM + kk + c);
      v4h ha; ha[0] = (_Float16)va.x; ha[1] = (_Float16)va.y;
              ha[2] = (_Float16)va.z; ha[3] = (_Float16)va.w;
      *(v4h*)&As[buf][r][c] = ha;
      float4 vb = *(const float4*)(W + (size_t)(bn + r) * K_DIM + kk + c);
      v4h hb; hb[0] = (_Float16)vb.x; hb[1] = (_Float16)vb.y;
              hb[2] = (_Float16)vb.z; hb[3] = (_Float16)vb.w;
      *(v4h*)&Bs[buf][r][c] = hb;
      if (kk + 2 * BK_H < K_DIM) {      // warm L2 for the K+2 tile
        __builtin_prefetch(X + (size_t)(bm + r) * K_DIM + kk + BK_H + c, 0, 3);
        __builtin_prefetch(W + (size_t)(bn + r) * K_DIM + kk + BK_H + c, 0, 3);
      }
    }
  };

  load_tiles(0, 0);
  __syncthreads();

  int buf = 0;
  for (int kk = 0; kk < K_DIM; kk += BK_H, buf ^= 1) {
    if (kk + BK_H < K_DIM) load_tiles(buf ^ 1, kk + BK_H);

    // A fragment: lanes 0-15 -> M rows with K {0..7,16..23}; lanes 16-31 -> K {8..15,24..31}
    v16h a[2], b[4];
#pragma unroll
    for (int fm = 0; fm < 2; ++fm) {
      const _Float16* base = &As[buf][wm * 32 + fm * 16 + l15][0];
      v8h lo = *(const v8h*)(base + hs * 8);
      v8h hi = *(const v8h*)(base + 16 + hs * 8);
#pragma unroll
      for (int i = 0; i < 8; ++i) { a[fm][i] = lo[i]; a[fm][i + 8] = hi[i]; }
    }
    // B fragment: lane holds one N column; lanes 0-15 K {0..7,16..23}, lanes 16-31 K {8..15,24..31}
#pragma unroll
    for (int fn = 0; fn < 4; ++fn) {
      const _Float16* base = &Bs[buf][wn * 64 + fn * 16 + l15][0];
      v8h lo = *(const v8h*)(base + hs * 8);
      v8h hi = *(const v8h*)(base + 16 + hs * 8);
#pragma unroll
      for (int i = 0; i < 8; ++i) { b[fn][i] = lo[i]; b[fn][i + 8] = hi[i]; }
    }
#pragma unroll
    for (int fm = 0; fm < 2; ++fm)
#pragma unroll
      for (int fn = 0; fn < 4; ++fn)
        acc[fm][fn] = __builtin_amdgcn_wmma_f32_16x16x32_f16(
            false, a[fm], false, b[fn], (short)0, acc[fm][fn], false, false);
    __syncthreads();
  }

  // C layout: VGPR r, lanes 0-15 -> M=r, lanes 16-31 -> M=r+8; N = lane&15
#pragma unroll
  for (int fm = 0; fm < 2; ++fm)
#pragma unroll
    for (int fn = 0; fn < 4; ++fn) {
      const int col = bn + wn * 64 + fn * 16 + l15;
      const float bv = bias[col];
#pragma unroll
      for (int r = 0; r < 8; ++r) {
        const int rowi = bm + wm * 32 + fm * 16 + hs * 8 + r;
        Y[(size_t)rowi * N_DIM + col] = acc[fm][fn][r] + bv;
      }
    }
}

// ============================ fp32 WMMA path ===============================
// Tile fill uses CDNA5 async global->LDS copies (no format conversion needed,
// bypasses VGPRs entirely, tracked by ASYNCcnt).
constexpr int BK_F  = 16;
constexpr int LDA_F = BK_F + 4;   // 20-float row stride

__global__ __launch_bounds__(256) void gemm_f32_wmma(const float* __restrict__ X,
                                                     const float* __restrict__ W,
                                                     const float* __restrict__ bias,
                                                     float* __restrict__ Y,
                                                     const int* __restrict__ flag) {
  if (flag[0] != 0) return;   // uniform across grid

  __shared__ alignas(16) float As[2][BM][LDA_F];
  __shared__ alignas(16) float Bs[2][BN][LDA_F];

  const int tid  = threadIdx.x;
  const int lane = tid & 31;
  const int wave = tid >> 5;
  const int wm   = wave & 3;
  const int wn   = wave >> 2;
  const int hs   = lane >> 4;
  const int l15  = lane & 15;
  const int bm   = blockIdx.x * BM;
  const int bn   = blockIdx.y * BN;

  v8f acc[2][4];
#pragma unroll
  for (int i = 0; i < 2; ++i)
#pragma unroll
    for (int j = 0; j < 4; ++j) acc[i][j] = v8f{};

  // Async tile fill: each lane copies 16B global -> LDS directly.
  // LDS offset = low 32 bits of the flat address (ISA 10.2 aperture mapping).
  auto load_tiles_async = [&](int buf, int kk) {
#pragma unroll
    for (int i = 0; i < 2; ++i) {
      int chunk = tid + i * 256;        // 512 float4 chunks per tile
      int r = chunk >> 2;               // 0..127
      int c = (chunk & 3) * 4;          // 0..12
      unsigned long long ga = (unsigned long long)(uintptr_t)(X + (size_t)(bm + r) * K_DIM + kk + c);
      unsigned int       la = (unsigned int)(uintptr_t)&As[buf][r][c];
      asm volatile("global_load_async_to_lds_b128 %0, %1, off"
                   :: "v"(la), "v"(ga) : "memory");
      unsigned long long gb = (unsigned long long)(uintptr_t)(W + (size_t)(bn + r) * K_DIM + kk + c);
      unsigned int       lb = (unsigned int)(uintptr_t)&Bs[buf][r][c];
      asm volatile("global_load_async_to_lds_b128 %0, %1, off"
                   :: "v"(lb), "v"(gb) : "memory");
    }
  };

  auto async_wait_barrier = [&]() {
    asm volatile("s_wait_asynccnt 0x0" ::: "memory");
    __syncthreads();
  };

  load_tiles_async(0, 0);
  async_wait_barrier();

  int buf = 0;
  for (int kk = 0; kk < K_DIM; kk += BK_F, buf ^= 1) {
    if (kk + BK_F < K_DIM) load_tiles_async(buf ^ 1, kk + BK_F);

#if __has_builtin(__builtin_amdgcn_wmma_f32_16x16x4_f32)
#pragma unroll
    for (int ks = 0; ks < BK_F / 4; ++ks) {
      // fp32 A 16x4: lanes 0-15 -> K {0,1}; lanes 16-31 -> K {2,3}
      v2f a[2], b[4];
#pragma unroll
      for (int fm = 0; fm < 2; ++fm)
        a[fm] = *(const v2f*)(&As[buf][wm * 32 + fm * 16 + l15][ks * 4 + hs * 2]);
#pragma unroll
      for (int fn = 0; fn < 4; ++fn)
        b[fn] = *(const v2f*)(&Bs[buf][wn * 64 + fn * 16 + l15][ks * 4 + hs * 2]);
#pragma unroll
      for (int fm = 0; fm < 2; ++fm)
#pragma unroll
        for (int fn = 0; fn < 4; ++fn)
          acc[fm][fn] = __builtin_amdgcn_wmma_f32_16x16x4_f32(
              false, a[fm], false, b[fn], (short)0, acc[fm][fn], false, false);
    }
#else
    // FMA fallback with identical per-lane C-element mapping
#pragma unroll
    for (int fm = 0; fm < 2; ++fm)
#pragma unroll
      for (int fn = 0; fn < 4; ++fn)
#pragma unroll
        for (int r = 0; r < 8; ++r) {
          const int mrow = wm * 32 + fm * 16 + hs * 8 + r;
          const int ncol = wn * 64 + fn * 16 + l15;
          float s = acc[fm][fn][r];
          for (int k = 0; k < BK_F; ++k)
            s = fmaf(As[buf][mrow][k], Bs[buf][ncol][k], s);
          acc[fm][fn][r] = s;
        }
#endif
    async_wait_barrier();   // waits loads issued this iter, then wg barrier
  }

#pragma unroll
  for (int fm = 0; fm < 2; ++fm)
#pragma unroll
    for (int fn = 0; fn < 4; ++fn) {
      const int col = bn + wn * 64 + fn * 16 + l15;
      const float bv = bias[col];
#pragma unroll
      for (int r = 0; r < 8; ++r) {
        const int rowi = bm + wm * 32 + fm * 16 + hs * 8 + r;
        Y[(size_t)rowi * N_DIM + col] = acc[fm][fn][r] + bv;
      }
    }
}

// ---------------------------------------------------------------------------
extern "C" void kernel_launch(void* const* d_in, const int* in_sizes, int n_in,
                              void* d_out, int out_size, void* d_ws, size_t ws_size,
                              hipStream_t stream) {
  (void)in_sizes; (void)n_in; (void)out_size; (void)ws_size;
  const float* x    = (const float*)d_in[0];   // [16384, 2048]
  const float* w    = (const float*)d_in[1];   // [512, 2048]
  const float* bias = (const float*)d_in[2];   // [512]
  float* y = (float*)d_out;                    // [16384, 512]

  float* wsf  = (float*)d_ws;        // ws[0]: sum of scaling factors
  int*   flag = (int*)d_ws + 1;      // ws[1]: precision flag

  ws_init_kernel<<<1, 1, 0, stream>>>(wsf);
  entropy_kernel<<<M_DIM, 256, 0, stream>>>(x, wsf);
  finalize_kernel<<<1, 1, 0, stream>>>(wsf, flag);

  dim3 grid(M_DIM / BM, N_DIM / BN);   // (128, 4)
  gemm_f16_wmma<<<grid, 256, 0, stream>>>(x, w, bias, y, flag);
  gemm_f32_wmma<<<grid, 256, 0, stream>>>(x, w, bias, y, flag);
}
